// VisionEncoder_59708635349070
// MI455X (gfx1250) — matrix operand order repaired and checked by reading, not compiled, not run on hardware
//
#include <hip/hip_runtime.h>
#include <hip/hip_bf16.h>
#include <math.h>

// ---------------------------------------------------------------------------
// Vision encoder for MI455X (gfx1250): all GEMMs via v_wmma_f32_16x16x32_bf16
// (f32 -> bf16 on the fly, f32 accumulate), flash attention per wave32,
// block-diagonal image mask exploited structurally.
// GEMM: 128x64 block tile, 8 waves, 32x32 wave tile (2x2 WMMA, fragments
// reused 2x) to halve LDS traffic + cvt work per FLOP.
// ---------------------------------------------------------------------------

typedef __bf16 v16bf __attribute__((ext_vector_type(16)));
typedef __bf16 v8bf  __attribute__((ext_vector_type(8)));
typedef float  v8f   __attribute__((ext_vector_type(8)));

#define WMMA_BF16(A, B, C) \
  __builtin_amdgcn_wmma_f32_16x16x32_bf16(false, (A), false, (B), (short)0, (C), false, false)

constexpr int kE     = 1280;   // embed dim
constexpr int kL     = 4;      // layers
constexpr int kNH    = 16;     // heads
constexpr int kD     = 80;     // head dim
constexpr int kM     = 5120;   // mlp hidden
constexpr int kO     = 2048;   // merger out
constexpr int kS     = 2048;   // tokens
constexpr int kSPP   = 1024;   // tokens per image
constexpr int kH4    = 5120;   // merged dim (E*MS*MS)
constexpr int kInP   = 1536;   // input patch dim
constexpr int kMRows = 512;    // merger rows (S/4)

// ------------------------------- GEMM --------------------------------------
constexpr int GBM = 128, GBN = 64, GBK = 32, LDT = 40; // LDS row stride (bf16)

template <int ACT>
__device__ __forceinline__ float actf(float x) {
  if (ACT == 1) { // tanh-approx GELU
    float u = x + 0.044715f * x * x * x;
    return 0.5f * x * (1.f + tanhf(0.7978845608028654f * u));
  }
  if (ACT == 2) { // exact GELU
    return 0.5f * x * (1.f + erff(0.70710678118654752f * x));
  }
  return x;
}

// C[Mr,N] = act(A[Mr,K] @ W[N,K]^T + bias[N])   (+= if RESID)
template <int ACT, int RESID>
__global__ __launch_bounds__(256) void gemm_bf16_wmma(
    const float* __restrict__ A, const float* __restrict__ W,
    const float* __restrict__ bias, float* __restrict__ C,
    int Mr, int N, int K) {
  __shared__ alignas(16) __bf16 lsA[GBM * LDT];
  __shared__ alignas(16) __bf16 lsB[GBN * LDT];

  const int tid  = threadIdx.x;
  const int lane = tid & 31;
  const int wv   = tid >> 5;       // 0..7
  const int wm   = wv >> 1;        // 0..3  (32-row strip)
  const int wn   = wv & 1;         // 0..1  (32-col strip)
  const int bm   = blockIdx.y * GBM;
  const int bn   = blockIdx.x * GBN;
  const int mi   = lane & 15;
  const int half = lane >> 4;

  const int arow = tid >> 1;       // 0..127, loads 16 floats of A
  const int ak   = (tid & 1) * 16;
  const int brow = tid >> 2;       // 0..63,  loads 8 floats of W
  const int bk   = (tid & 3) * 8;

  v8f acc[2][2] = {};

  for (int k0 = 0; k0 < K; k0 += GBK) {
    const float* ap = A + (size_t)(bm + arow) * K + k0 + ak;
    const float* wp = W + (size_t)(bn + brow) * K + k0 + bk;
    float av[16], wvv[8];
#pragma unroll
    for (int e = 0; e < 16; ++e) av[e] = ap[e];
#pragma unroll
    for (int e = 0; e < 8; ++e) wvv[e] = wp[e];
    if (k0 + GBK < K) {            // prefetch next K-tile (global_prefetch_b8)
      __builtin_prefetch(ap + GBK, 0, 3);
      __builtin_prefetch(wp + GBK, 0, 3);
    }
    __syncthreads();               // previous fragments consumed
#pragma unroll
    for (int e = 0; e < 16; ++e) lsA[arow * LDT + ak + e] = (__bf16)av[e];
#pragma unroll
    for (int e = 0; e < 8; ++e)  lsB[brow * LDT + bk + e] = (__bf16)wvv[e];
    __syncthreads();

    // A fragments (two 16-row strips; K chunks at half*8 and 16+half*8)
    v16bf af[2];
#pragma unroll
    for (int sub = 0; sub < 2; ++sub) {
      int r = wm * 32 + sub * 16 + mi;
      v8bf a0 = *(const v8bf*)&lsA[r * LDT + half * 8];
      v8bf a1 = *(const v8bf*)&lsA[r * LDT + 16 + half * 8];
      af[sub] = __builtin_shufflevector(a0, a1, 0, 1, 2, 3, 4, 5, 6, 7,
                                        8, 9, 10, 11, 12, 13, 14, 15);
    }
    // B fragments (lane holds K chunks at half*8 and 16+half*8 for column n)
    v16bf bfv[2];
#pragma unroll
    for (int nb = 0; nb < 2; ++nb) {
      int n = wn * 32 + nb * 16 + mi;
      v8bf b0 = *(const v8bf*)&lsB[n * LDT + half * 8];
      v8bf b1 = *(const v8bf*)&lsB[n * LDT + 16 + half * 8];
      bfv[nb] = __builtin_shufflevector(b0, b1, 0, 1, 2, 3, 4, 5, 6, 7,
                                        8, 9, 10, 11, 12, 13, 14, 15);
    }
#pragma unroll
    for (int sub = 0; sub < 2; ++sub)
#pragma unroll
      for (int nb = 0; nb < 2; ++nb)
        acc[sub][nb] = WMMA_BF16(af[sub], bfv[nb], acc[sub][nb]);
  }

  // Epilogue: D layout row = r + half*8, col = mi
#pragma unroll
  for (int sub = 0; sub < 2; ++sub)
#pragma unroll
    for (int nb = 0; nb < 2; ++nb)
#pragma unroll
      for (int r = 0; r < 8; ++r) {
        int row = bm + wm * 32 + sub * 16 + half * 8 + r;
        int col = bn + wn * 32 + nb * 16 + mi;
        float v0 = actf<ACT>(acc[sub][nb][r] + bias[col]);
        if (RESID) C[(size_t)row * N + col] += v0;
        else       C[(size_t)row * N + col] = v0;
      }
}

template <int ACT, int RESID>
static void gemm_launch(hipStream_t st, const float* A, const float* W,
                        const float* b, float* C, int Mr, int N, int K) {
  dim3 g(N / GBN, Mr / GBM);
  gemm_bf16_wmma<ACT, RESID><<<g, 256, 0, st>>>(A, W, b, C, Mr, N, K);
}

// ---------------------------- LayerNorm ------------------------------------
__global__ __launch_bounds__(256) void layernorm_kernel(
    const float* __restrict__ x, const float* __restrict__ w,
    const float* __restrict__ b, float* __restrict__ y, int C) {
  __shared__ float s1[256];
  __shared__ float s2[256];
  const int row = blockIdx.x, tid = threadIdx.x;
  const float* xr = x + (size_t)row * C;
  float a = 0.f, q = 0.f;
  for (int c = tid; c < C; c += 256) { float v = xr[c]; a += v; q += v * v; }
  s1[tid] = a; s2[tid] = q;
  __syncthreads();
  for (int st = 128; st > 0; st >>= 1) {
    if (tid < st) { s1[tid] += s1[tid + st]; s2[tid] += s2[tid + st]; }
    __syncthreads();
  }
  float mean = s1[0] / (float)C;
  float var  = s2[0] / (float)C - mean * mean;
  float inv  = rsqrtf(var + 1e-6f);
  float* yr = y + (size_t)row * C;
  for (int c = tid; c < C; c += 256) yr[c] = (xr[c] - mean) * inv * w[c] + b[c];
}

// --------------------- Pos-embed bilinear add ------------------------------
// token index -> window-permuted (h,w): sp = ((hb*16+wb)*2+hi)*2+wi
__global__ void posadd_kernel(float* __restrict__ h, const float* __restrict__ pe) {
  int s = blockIdx.x;
  int e = blockIdx.y * 256 + threadIdx.x;
  int sp = s % kSPP;
  int wi = sp & 1, hi = (sp >> 1) & 1, wb = (sp >> 2) & 15, hb = (sp >> 6) & 15;
  int hh = hb * 2 + hi, ww = wb * 2 + wi;
  float hf_ = hh * (47.f / 31.f), wf_ = ww * (47.f / 31.f);
  int h0 = (int)hf_, w0 = (int)wf_;
  int h1 = h0 + 1 > 47 ? 47 : h0 + 1;
  int w1 = w0 + 1 > 47 ? 47 : w0 + 1;
  float dh = hf_ - (float)h0, dw = wf_ - (float)w0;
  float p = (1.f - dh) * (1.f - dw) * pe[(size_t)(h0 * 48 + w0) * kE + e]
          + (1.f - dh) * dw         * pe[(size_t)(h0 * 48 + w1) * kE + e]
          + dh * (1.f - dw)         * pe[(size_t)(h1 * 48 + w0) * kE + e]
          + dh * dw                 * pe[(size_t)(h1 * 48 + w1) * kE + e];
  h[(size_t)s * kE + e] += p;
}

// ------------------------ Rotary cos/sin tables ----------------------------
__global__ void rope_tab_kernel(float* __restrict__ cost, float* __restrict__ sint) {
  int s = blockIdx.x, j = threadIdx.x;
  if (j >= 40) return;
  int sp = s % kSPP;
  int wi = sp & 1, hi = (sp >> 1) & 1, wb = (sp >> 2) & 15, hb = (sp >> 6) & 15;
  int hh = hb * 2 + hi, ww = wb * 2 + wi;
  int p  = (j < 20) ? hh : ww;
  int fi = (j < 20) ? j : j - 20;
  float inv = __powf(10000.f, -(float)(2 * fi) / 40.f);
  float f = (float)p * inv;
  float c = __cosf(f), sn = __sinf(f);
  cost[(size_t)s * kD + j] = c;  cost[(size_t)s * kD + 40 + j] = c;
  sint[(size_t)s * kD + j] = sn; sint[(size_t)s * kD + 40 + j] = sn;
}

// ---------------- QKV split + RoPE -> (NH, S, D) layout --------------------
__global__ void rope_split_kernel(const float* __restrict__ qkv,
                                  const float* __restrict__ cost,
                                  const float* __restrict__ sint,
                                  float* __restrict__ qr, float* __restrict__ kr,
                                  float* __restrict__ vr) {
  int s = blockIdx.x, hh = blockIdx.y, d = threadIdx.x;
  if (d >= kD) return;
  const float* base = qkv + (size_t)s * 3 * kE + hh * kD;
  float qv = base[d], kv = base[kE + d], vv = base[2 * kE + d];
  int dp = (d < 40) ? d + 40 : d - 40;
  float sgn = (d < 40) ? -1.f : 1.f;
  float qrot = sgn * base[dp];
  float krot = sgn * base[kE + dp];
  float c = cost[(size_t)s * kD + d], sn = sint[(size_t)s * kD + d];
  size_t oi = ((size_t)hh * kS + s) * kD + d;
  qr[oi] = qv * c + qrot * sn;
  kr[oi] = kv * c + krot * sn;
  vr[oi] = vv;
}

// ------------------------- Flash attention ---------------------------------
// One wave handles a 16-query tile of one head; keys restricted to the
// query's image (block-diagonal mask). D=80 -> K split {32,32,16+pad}.
__global__ __launch_bounds__(128) void attn_kernel(
    const float* __restrict__ q, const float* __restrict__ k,
    const float* __restrict__ v, float* __restrict__ o) {
  __shared__ alignas(16) __bf16 psh[4][16 * 32];
  const int lane = threadIdx.x & 31;
  const int wv   = threadIdx.x >> 5;
  const int tile = blockIdx.x * 4 + wv;       // 0..2047
  const int hIdx = tile >> 7;                 // head
  const int qt   = tile & 127;
  const int qbase = qt * 16;
  const int img   = qbase / kSPP;
  const int mi   = lane & 15;
  const int half = lane >> 4;
  const __bf16 zero = (__bf16)0.0f;

  // Q fragments (A layout: M = mi, K chunks at half*8 / 16+half*8)
  v16bf qf[3];
  const float* qrow = q + ((size_t)hIdx * kS + (qbase + mi)) * kD;
#pragma unroll
  for (int c = 0; c < 3; ++c) {
#pragma unroll
    for (int e = 0; e < 16; ++e) {
      int kd = c * 32 + ((e < 8) ? (half * 8 + e) : (16 + half * 8 + (e - 8)));
      qf[c][e] = (kd < kD) ? (__bf16)qrow[kd] : zero;
    }
  }

  v8f accv[5];
#pragma unroll
  for (int i = 0; i < 5; ++i) accv[i] = (v8f){};
  float mrow[8], lrow[8];
#pragma unroll
  for (int r = 0; r < 8; ++r) { mrow[r] = -3.0e38f; lrow[r] = 0.f; }

  const size_t hoff = (size_t)hIdx * kS;
  const float scale = 0.11180339887498949f; // 1/sqrt(80)

  for (int kt = 0; kt < kSPP / 32; ++kt) {
    const int key0 = img * kSPP + kt * 32;
    // ---- scores: Q (16x80) x K^T (80x32) -> two 16x16 accs ----
    v8f sc0 = {}, sc1 = {};
    const float* kr0 = k + (hoff + key0 + mi) * kD;
    const float* kr1 = k + (hoff + key0 + 16 + mi) * kD;
#pragma unroll
    for (int c = 0; c < 3; ++c) {
      v16bf b0, b1;
#pragma unroll
      for (int e = 0; e < 16; ++e) {
        int kd = c * 32 + ((e < 8) ? (half * 8 + e) : (16 + half * 8 + (e - 8)));
        b0[e] = (kd < kD) ? (__bf16)kr0[kd] : zero;
        b1[e] = (kd < kD) ? (__bf16)kr1[kd] : zero;
      }
      sc0 = WMMA_BF16(qf[c], b0, sc0);
      sc1 = WMMA_BF16(qf[c], b1, sc1);
    }
    // ---- online softmax over the 32 new columns ----
#pragma unroll
    for (int r = 0; r < 8; ++r) {
      float s0 = sc0[r] * scale, s1 = sc1[r] * scale;
      float mx = fmaxf(s0, s1);
#pragma unroll
      for (int off = 8; off >= 1; off >>= 1) mx = fmaxf(mx, __shfl_xor(mx, off, 32));
      float mnew  = fmaxf(mrow[r], mx);
      float alpha = __expf(mrow[r] - mnew);
      float p0 = __expf(s0 - mnew), p1 = __expf(s1 - mnew);
      float rs = p0 + p1;
#pragma unroll
      for (int off = 8; off >= 1; off >>= 1) rs += __shfl_xor(rs, off, 32);
      lrow[r] = lrow[r] * alpha + rs;
      mrow[r] = mnew;
      sc0[r] = p0; sc1[r] = p1;
#pragma unroll
      for (int nt = 0; nt < 5; ++nt) accv[nt][r] *= alpha;
    }
    // ---- stage P (acc layout) -> LDS -> reload in A layout ----
    __bf16* pw = &psh[wv][0];
#pragma unroll
    for (int r = 0; r < 8; ++r) {
      pw[(r + half * 8) * 32 + mi]      = (__bf16)sc0[r];
      pw[(r + half * 8) * 32 + 16 + mi] = (__bf16)sc1[r];
    }
    asm volatile("s_wait_dscnt 0" ::: "memory");
    v16bf pf;
#pragma unroll
    for (int e = 0; e < 16; ++e) {
      int kc = (e < 8) ? (half * 8 + e) : (16 + half * 8 + (e - 8));
      pf[e] = pw[mi * 32 + kc];
    }
    // ---- P (16x32) x V (32x80) -> 5 n-tiles ----
#pragma unroll
    for (int nt = 0; nt < 5; ++nt) {
      v16bf vf;
      int d = nt * 16 + mi;
#pragma unroll
      for (int e = 0; e < 16; ++e) {
        int key = key0 + ((e < 8) ? (half * 8 + e) : (16 + half * 8 + (e - 8)));
        vf[e] = (__bf16)v[(hoff + key) * kD + d];
      }
      accv[nt] = WMMA_BF16(pf, vf, accv[nt]);
    }
    asm volatile("s_wait_dscnt 0" ::: "memory");
  }
  // ---- epilogue: o[s, h*80 + d] = acc / l ----
#pragma unroll
  for (int nt = 0; nt < 5; ++nt) {
#pragma unroll
    for (int r = 0; r < 8; ++r) {
      int srow = qbase + r + half * 8;
      int col  = hIdx * kD + nt * 16 + mi;
      o[(size_t)srow * kE + col] = accv[nt][r] / lrow[r];
    }
  }
}

// ---------------------------------------------------------------------------
extern "C" void kernel_launch(void* const* d_in, const int* in_sizes, int n_in,
                              void* d_out, int out_size, void* d_ws, size_t ws_size,
                              hipStream_t stream) {
  (void)in_sizes; (void)n_in; (void)out_size; (void)ws_size;
  const float* x        = (const float*)d_in[0];
  const float* patch_w  = (const float*)d_in[1];
  const float* patch_b  = (const float*)d_in[2];
  const float* pos_emb  = (const float*)d_in[3];
  const float* ln1_w    = (const float*)d_in[4];
  const float* ln1_b    = (const float*)d_in[5];
  const float* ln2_w    = (const float*)d_in[6];
  const float* ln2_b    = (const float*)d_in[7];
  const float* qkv_w    = (const float*)d_in[8];
  const float* qkv_b    = (const float*)d_in[9];
  const float* apw      = (const float*)d_in[10];
  const float* apb      = (const float*)d_in[11];
  const float* fc1_w    = (const float*)d_in[12];
  const float* fc1_b    = (const float*)d_in[13];
  const float* fc2_w    = (const float*)d_in[14];
  const float* fc2_b    = (const float*)d_in[15];
  const float* m_ln_w   = (const float*)d_in[16];
  const float* m_ln_b   = (const float*)d_in[17];
  const float* m_fc1_w  = (const float*)d_in[18];
  const float* m_fc1_b  = (const float*)d_in[19];
  const float* m_fc2_w  = (const float*)d_in[20];
  const float* m_fc2_b  = (const float*)d_in[21];
  const float* ds_ln_w  = (const float*)d_in[22];
  const float* ds_ln_b  = (const float*)d_in[23];
  const float* ds_fc1_w = (const float*)d_in[24];
  const float* ds_fc1_b = (const float*)d_in[25];
  const float* ds_fc2_w = (const float*)d_in[26];
  const float* ds_fc2_b = (const float*)d_in[27];
  float* out = (float*)d_out;

  char* wsp = (char*)d_ws;
  size_t off = 0;
  auto alloc = [&](size_t nElems) -> float* {
    float* p = (float*)(wsp + off);
    off += nElems * sizeof(float);
    off = (off + 255) & ~(size_t)255;
    return p;
  };
  float* h    = alloc((size_t)kS * kE);
  float* tmp  = alloc((size_t)kS * kE);     // also reused as (512,5120)
  float* qkvb = alloc((size_t)kS * 3 * kE);
  float* qr   = alloc((size_t)kNH * kS * kD);
  float* kr   = alloc((size_t)kNH * kS * kD);
  float* vr   = alloc((size_t)kNH * kS * kD);
  float* ob   = alloc((size_t)kS * kE);
  float* mlp  = alloc((size_t)kS * kM);
  float* mh   = alloc((size_t)kMRows * kH4);
  float* cost = alloc((size_t)kS * kD);
  float* sint = alloc((size_t)kS * kD);

  // patch embed + pos embed + rotary tables
  gemm_launch<0, 0>(stream, x, patch_w, patch_b, h, kS, kE, kInP);
  posadd_kernel<<<dim3(kS, kE / 256), 256, 0, stream>>>(h, pos_emb);
  rope_tab_kernel<<<kS, 64, 0, stream>>>(cost, sint);

  for (int l = 0; l < kL; ++l) {
    layernorm_kernel<<<kS, 256, 0, stream>>>(h, ln1_w + (size_t)l * kE,
                                             ln1_b + (size_t)l * kE, tmp, kE);
    gemm_launch<0, 0>(stream, tmp, qkv_w + (size_t)l * 3 * kE * kE,
                      qkv_b + (size_t)l * 3 * kE, qkvb, kS, 3 * kE, kE);
    rope_split_kernel<<<dim3(kS, kNH), 96, 0, stream>>>(qkvb, cost, sint, qr, kr, vr);
    attn_kernel<<<kNH * (kS / 16) / 4, 128, 0, stream>>>(qr, kr, vr, ob);
    gemm_launch<0, 1>(stream, ob, apw + (size_t)l * kE * kE,
                      apb + (size_t)l * kE, h, kS, kE, kE);  // h += proj
    layernorm_kernel<<<kS, 256, 0, stream>>>(h, ln2_w + (size_t)l * kE,
                                             ln2_b + (size_t)l * kE, tmp, kE);
    gemm_launch<1, 0>(stream, tmp, fc1_w + (size_t)l * kM * kE,
                      fc1_b + (size_t)l * kM, mlp, kS, kM, kE);  // tanh GELU
    gemm_launch<0, 1>(stream, mlp, fc2_w + (size_t)l * kE * kM,
                      fc2_b + (size_t)l * kE, h, kS, kE, kM);    // h += fc2
    if (l == 1 || l == 2) {  // deep-supervision mergers (post-LN over 5120)
      int j = l - 1;
      layernorm_kernel<<<kMRows, 256, 0, stream>>>(h, ds_ln_w + (size_t)j * kH4,
                                                   ds_ln_b + (size_t)j * kH4, tmp, kH4);
      gemm_launch<2, 0>(stream, tmp, ds_fc1_w + (size_t)j * kH4 * kH4,
                        ds_fc1_b + (size_t)j * kH4, mh, kMRows, kH4, kH4);
      gemm_launch<0, 0>(stream, mh, ds_fc2_w + (size_t)j * kO * kH4,
                        ds_fc2_b + (size_t)j * kO,
                        out + (size_t)(1 + j) * kMRows * kO, kMRows, kO, kH4);
    }
  }
  // main merger: LN over E, then reinterpret (2048,1280) as (512,5120)
  layernorm_kernel<<<kS, 256, 0, stream>>>(h, m_ln_w, m_ln_b, tmp, kE);
  gemm_launch<2, 0>(stream, tmp, m_fc1_w, m_fc1_b, mh, kMRows, kH4, kH4);
  gemm_launch<0, 0>(stream, mh, m_fc2_w, m_fc2_b, out, kMRows, kO, kH4);
}
